// EcoGAT_31542239822515
// MI455X (gfx1250) — matrix operand order repaired and checked by reading, not compiled, not run on hardware
//
#include <hip/hip_runtime.h>
#include <hip/hip_bf16.h>
#include <math.h>

typedef __attribute__((ext_vector_type(16))) _Float16 v16h;
typedef __attribute__((ext_vector_type(8)))  float    v8f;

#define NODES 40000
#define EDGES 640000
#define EPR   (EDGES + NODES)   /* 680000 edges incl. self loops */
#define HEADS 8
#define CH    16
#define HIDC  128
#define FIN   500

/* d_out layout (float elements): z, recon, alpha1, alpha2 */
#define OUT_Z      0
#define OUT_RECON  (NODES*HIDC)                 /* 5,120,000  */
#define OUT_A1     (OUT_RECON + NODES*FIN)      /* 25,120,000 */
#define OUT_A2     (OUT_A1 + EPR*HEADS)         /* 30,560,000 */

/* workspace layout (float elements) */
#define WS_H    0                     /* [N,128] raw GEMM out (layer1 then layer2) */
#define WS_HN   (NODES*HIDC)          /* [N,128] normalized layer-1 output         */
#define WS_AGG  (2*NODES*HIDC)        /* [N,128] scatter accumulator               */
#define WS_AS   (3*NODES*HIDC)
#define WS_AD   (WS_AS + NODES*HEADS)
#define WS_EMAX (WS_AD + NODES*HEADS)
#define WS_ESUM (WS_EMAX + NODES*HEADS)

__device__ inline unsigned int pack2h(float v0, float v1) {
  _Float16 h0 = (_Float16)v0, h1 = (_Float16)v1;
  unsigned short u0 = __builtin_bit_cast(unsigned short, h0);
  unsigned short u1 = __builtin_bit_cast(unsigned short, h1);
  return (unsigned int)u0 | ((unsigned int)u1 << 16);
}

/* fragment-major LDS offsets (in uints):
 * A element (row 0..63, k 0..31):
 *   frag = row>>4, lane = (row&15)+16*((k>>3)&1), idx = (k&7)+8*(k>>4)
 * B element (k 0..31, col 0..127):
 *   frag = col>>4, lane = (col&15)+16*((k>>3)&1), idx = (k&7)+8*(k>>4)
 */

/* ----------------------------------------------------------------------
 * Tiled WMMA GEMM:  C[M,N] = A[M,K] @ B[K,N] (+ bias[N])
 * Block = 256 threads = 8 waves. Tile 64(M) x 128(N), BK=32.
 * Wave grid 2(Mw) x 4(Nw); each wave owns 32x32 = 4 WMMAs per K-step.
 * Fragment-major LDS: each lane's v16h fragment = one 32B chunk
 * (2x ds_load_b128). K-tail peeled out of the hot loop; N-mask branch
 * hoisted (block-uniform). M must be a multiple of 64 (40000 = 625*64).
 * -------------------------------------------------------------------- */
__global__ __launch_bounds__(256)
void gemm_wmma_f16(const float* __restrict__ A, const float* __restrict__ B,
                   const float* __restrict__ bias, float* __restrict__ C,
                   int M, int N, int K, int lda, int ldb, int ldc, int addBias) {
  __shared__ __align__(32) unsigned int AfragU[4 * 32 * 8];   /* 4KB */
  __shared__ __align__(32) unsigned int BfragU[8 * 32 * 8];   /* 8KB */

  const int t    = threadIdx.x;
  const int lane = t & 31;
  const int wave = t >> 5;           /* 0..7 */
  const int wm   = wave >> 2;        /* 0..1 : 32-row slice  */
  const int wn   = wave & 3;         /* 0..3 : 32-col slice  */
  const int mBase = blockIdx.y * 64;
  const int nBase = blockIdx.x * 128;
  const bool nMask = (nBase + 128 > N);

  /* per-thread staging coordinates (loop-invariant) */
  const int aRow = t >> 4;            /* 0..15 per r-step stride 16 rows */
  const int aC2  = (t & 15) * 2;      /* even k */
  const int bCol = t & 127;           /* 0..127 */
  const int bK2  = (t >> 7) * 2;      /* 0 or 2, +4 per r-step */

  /* LDS write offsets (uints), invariant per r */
  int aOffs[4], bOffs[8];
#pragma unroll
  for (int r = 0; r < 4; ++r) {
    int row = aRow + r * 16;
    int f   = row >> 4;
    int ln  = (row & 15) + 16 * ((aC2 >> 3) & 1);
    int idx = (aC2 & 7) + 8 * (aC2 >> 4);
    aOffs[r] = (f * 32 + ln) * 8 + (idx >> 1);
  }
#pragma unroll
  for (int r = 0; r < 8; ++r) {
    int k2  = bK2 + r * 4;
    int f   = bCol >> 4;
    int ln  = (bCol & 15) + 16 * ((k2 >> 3) & 1);
    int idx = (k2 & 7) + 8 * (k2 >> 4);
    bOffs[r] = (f * 32 + ln) * 8 + (idx >> 1);
  }

  v8f acc00 = {}, acc01 = {}, acc10 = {}, acc11 = {};

  const int aFrag0 = ((2 * wm + 0) * 32 + lane) * 8;
  const int aFrag1 = ((2 * wm + 1) * 32 + lane) * 8;
  const int bFrag0 = ((2 * wn + 0) * 32 + lane) * 8;
  const int bFrag1 = ((2 * wn + 1) * 32 + lane) * 8;

  const int kFull = K & ~31;

  /* =================== hot loop: full K-steps, no masks =================== */
  for (int kBase = 0; kBase < kFull; kBase += 32) {
    /* stage A 64x32 (pairs -> b64 global loads, b32 LDS stores) */
#pragma unroll
    for (int r = 0; r < 4; ++r) {
      const float* p = &A[(size_t)(mBase + aRow + r * 16) * lda + kBase + aC2];
      AfragU[aOffs[r]] = pack2h(p[0], p[1]);
    }
    /* stage B 32x128 */
    if (!nMask) {
#pragma unroll
      for (int r = 0; r < 8; ++r) {
        int k = kBase + bK2 + r * 4;
        float v0 = B[(size_t)k * ldb + nBase + bCol];
        float v1 = B[(size_t)(k + 1) * ldb + nBase + bCol];
        BfragU[bOffs[r]] = pack2h(v0, v1);
      }
    } else {
      int ng  = nBase + bCol;
      int ngc = (ng < N) ? ng : N - 1;
#pragma unroll
      for (int r = 0; r < 8; ++r) {
        int k = kBase + bK2 + r * 4;
        float v0 = B[(size_t)k * ldb + ngc];
        float v1 = B[(size_t)(k + 1) * ldb + ngc];
        if (ng >= N) { v0 = 0.0f; v1 = 0.0f; }
        BfragU[bOffs[r]] = pack2h(v0, v1);
      }
    }
    if (kBase + 32 < K)
      __builtin_prefetch(&A[(size_t)(mBase + (t >> 5) * 8) * lda + kBase + 32 + (t & 31)], 0, 0);
    __syncthreads();

    v16h a0 = *(const v16h*)&AfragU[aFrag0];
    v16h a1 = *(const v16h*)&AfragU[aFrag1];
    v16h b0 = *(const v16h*)&BfragU[bFrag0];
    v16h b1 = *(const v16h*)&BfragU[bFrag1];
    acc00 = __builtin_amdgcn_wmma_f32_16x16x32_f16(false, a0, false, b0, (short)0, acc00, false, false);
    acc01 = __builtin_amdgcn_wmma_f32_16x16x32_f16(false, a0, false, b1, (short)0, acc01, false, false);
    acc10 = __builtin_amdgcn_wmma_f32_16x16x32_f16(false, a1, false, b0, (short)0, acc10, false, false);
    acc11 = __builtin_amdgcn_wmma_f32_16x16x32_f16(false, a1, false, b1, (short)0, acc11, false, false);
    __syncthreads();
  }

  /* =================== peeled masked K-tail (runs at most once) =================== */
  if (kFull < K) {
    const int kBase = kFull;
#pragma unroll
    for (int r = 0; r < 4; ++r) {
      int k0 = kBase + aC2, k1 = k0 + 1;
      int k0c = (k0 < K) ? k0 : K - 1;
      int k1c = (k1 < K) ? k1 : K - 1;
      const float* row = &A[(size_t)(mBase + aRow + r * 16) * lda];
      float v0 = row[k0c]; if (k0 >= K) v0 = 0.0f;
      float v1 = row[k1c]; if (k1 >= K) v1 = 0.0f;
      AfragU[aOffs[r]] = pack2h(v0, v1);
    }
    {
      int ng  = nBase + bCol;
      int ngc = (ng < N) ? ng : N - 1;
#pragma unroll
      for (int r = 0; r < 8; ++r) {
        int k0 = kBase + bK2 + r * 4, k1 = k0 + 1;
        int k0c = (k0 < K) ? k0 : K - 1;
        int k1c = (k1 < K) ? k1 : K - 1;
        float v0 = B[(size_t)k0c * ldb + ngc]; if (k0 >= K || ng >= N) v0 = 0.0f;
        float v1 = B[(size_t)k1c * ldb + ngc]; if (k1 >= K || ng >= N) v1 = 0.0f;
        BfragU[bOffs[r]] = pack2h(v0, v1);
      }
    }
    __syncthreads();
    v16h a0 = *(const v16h*)&AfragU[aFrag0];
    v16h a1 = *(const v16h*)&AfragU[aFrag1];
    v16h b0 = *(const v16h*)&BfragU[bFrag0];
    v16h b1 = *(const v16h*)&BfragU[bFrag1];
    acc00 = __builtin_amdgcn_wmma_f32_16x16x32_f16(false, a0, false, b0, (short)0, acc00, false, false);
    acc01 = __builtin_amdgcn_wmma_f32_16x16x32_f16(false, a0, false, b1, (short)0, acc01, false, false);
    acc10 = __builtin_amdgcn_wmma_f32_16x16x32_f16(false, a1, false, b0, (short)0, acc10, false, false);
    acc11 = __builtin_amdgcn_wmma_f32_16x16x32_f16(false, a1, false, b1, (short)0, acc11, false, false);
    __syncthreads();
  }

  /* ---- store D (M always in range; mask N only on edge tiles) ---- */
  const int r0 = mBase + wm * 32 + (lane >> 4) * 8;
  const int c0 = nBase + wn * 32 + (lane & 15);
  const float bb0 = (addBias && c0      < N) ? bias[c0]      : 0.0f;
  const float bb1 = (addBias && c0 + 16 < N) ? bias[c0 + 16] : 0.0f;
#pragma unroll
  for (int j = 0; j < 8; ++j) {
    int ra = r0 + j, rb = r0 + 16 + j;
    if (!nMask || c0 < N) {
      C[(size_t)ra * ldc + c0] = acc00[j] + bb0;
      C[(size_t)rb * ldc + c0] = acc10[j] + bb0;
    }
    if (!nMask || c0 + 16 < N) {
      C[(size_t)ra * ldc + c0 + 16] = acc01[j] + bb1;
      C[(size_t)rb * ldc + c0 + 16] = acc11[j] + bb1;
    }
  }
}

/* a_s[n,h] = sum_c h[n,h,c]*att_s[h,c] ; same for a_d */
__global__ void attn_proj(const float* __restrict__ h, const float* __restrict__ att_s,
                          const float* __restrict__ att_d, float* __restrict__ as_,
                          float* __restrict__ ad_) {
  int gid = blockIdx.x * blockDim.x + threadIdx.x;
  if (gid >= NODES * HEADS) return;
  int n = gid >> 3, hd = gid & 7;
  const float* hp = h + (size_t)n * HIDC + hd * CH;
  const float* sp = att_s + hd * CH;
  const float* dp = att_d + hd * CH;
  float ss = 0.f, sd = 0.f;
#pragma unroll
  for (int c = 0; c < CH; ++c) { float v = hp[c]; ss += v * sp[c]; sd += v * dp[c]; }
  as_[gid] = ss; ad_[gid] = sd;
}

__global__ void seg_init(float* __restrict__ emax, float* __restrict__ esum,
                         float* __restrict__ agg) {
  int gid = blockIdx.x * blockDim.x + threadIdx.x;
  if (gid < NODES * HIDC) agg[gid] = 0.0f;
  if (gid < NODES * HEADS) { emax[gid] = -__builtin_inff(); esum[gid] = 0.0f; }
}

__device__ inline void atomicMaxF(float* addr, float val) {
  if (val >= 0.0f) atomicMax((int*)addr, __float_as_int(val));
  else             atomicMin((unsigned int*)addr, (unsigned int)__float_as_int(val));
}

__device__ inline void edge_sd(int e, const int* __restrict__ ei, int& s, int& d) {
  if (e < EDGES) { s = ei[e]; d = ei[EDGES + e]; }
  else           { s = e - EDGES; d = s; }
}

/* pass 1: e = leakyrelu(a_s[src]+a_d[dst], 0.2); store raw; segment max */
__global__ void edge_logits(const int* __restrict__ ei, const float* __restrict__ as_,
                            const float* __restrict__ ad_, float* __restrict__ emax,
                            float* __restrict__ alphaOut) {
  int gid = blockIdx.x * blockDim.x + threadIdx.x;
  if (gid >= EPR * HEADS) return;
  int e = gid >> 3, hd = gid & 7, s, d;
  edge_sd(e, ei, s, d);
  float v = as_[s * HEADS + hd] + ad_[d * HEADS + hd];
  v = (v > 0.0f) ? v : 0.2f * v;
  alphaOut[gid] = v;
  atomicMaxF(&emax[d * HEADS + hd], v);
}

/* pass 2: w = exp(e - max); segment sum */
__global__ void edge_exp(const int* __restrict__ ei, const float* __restrict__ emax,
                         float* __restrict__ esum, float* __restrict__ alphaOut) {
  int gid = blockIdx.x * blockDim.x + threadIdx.x;
  if (gid >= EPR * HEADS) return;
  int e = gid >> 3, hd = gid & 7, s, d;
  edge_sd(e, ei, s, d);
  float w = __expf(alphaOut[gid] - emax[d * HEADS + hd]);
  alphaOut[gid] = w;
  atomicAdd(&esum[d * HEADS + hd], w);
}

/* pass 3: alpha = w / (sum + 1e-16) — normalized in place in d_out */
__global__ void alpha_norm(const int* __restrict__ ei, const float* __restrict__ esum,
                           float* __restrict__ alphaOut) {
  int gid = blockIdx.x * blockDim.x + threadIdx.x;
  if (gid >= EPR * HEADS) return;
  int e = gid >> 3, hd = gid & 7, s, d;
  edge_sd(e, ei, s, d);
  alphaOut[gid] = alphaOut[gid] / (esum[d * HEADS + hd] + 1e-16f);
}

/* pass 4: agg[dst, ch] += alpha[e, ch/16] * h[src, ch] */
__global__ void edge_msg(const int* __restrict__ ei, const float* __restrict__ h,
                         const float* __restrict__ alpha, float* __restrict__ agg) {
  int gid = blockIdx.x * blockDim.x + threadIdx.x;
  if (gid >= EPR * HIDC) return;
  int e = gid >> 7, ch = gid & 127, s, d;
  edge_sd(e, ei, s, d);
  float a = alpha[e * HEADS + (ch >> 4)];
  atomicAdd(&agg[(size_t)d * HIDC + ch], a * h[(size_t)s * HIDC + ch]);
}

/* per-node: +bias, leakyrelu(0.01), LayerNorm (wave32 reduce), optional sigmoid */
__global__ __launch_bounds__(256)
void node_finalize(const float* __restrict__ agg, const float* __restrict__ bias,
                   const float* __restrict__ g, const float* __restrict__ beta,
                   float* __restrict__ out, int doSigmoid) {
  int lane = threadIdx.x & 31;
  int node = blockIdx.x * 8 + (threadIdx.x >> 5);
  if (node >= NODES) return;
  int cb = lane * 4;
  float v[4];
#pragma unroll
  for (int i = 0; i < 4; ++i) {
    float x = agg[(size_t)node * HIDC + cb + i] + bias[cb + i];
    v[i] = (x > 0.0f) ? x : 0.01f * x;
  }
  float s = v[0] + v[1] + v[2] + v[3];
  float q = v[0]*v[0] + v[1]*v[1] + v[2]*v[2] + v[3]*v[3];
#pragma unroll
  for (int off = 16; off > 0; off >>= 1) {
    s += __shfl_xor(s, off, 32);
    q += __shfl_xor(q, off, 32);
  }
  float mean = s * (1.0f / HIDC);
  float var  = q * (1.0f / HIDC) - mean * mean;
  float rs   = rsqrtf(var + 1e-5f);
#pragma unroll
  for (int i = 0; i < 4; ++i) {
    float y = (v[i] - mean) * rs * g[cb + i] + beta[cb + i];
    if (doSigmoid) y = 1.0f / (1.0f + __expf(-y));
    out[(size_t)node * HIDC + cb + i] = y;
  }
}

extern "C" void kernel_launch(void* const* d_in, const int* in_sizes, int n_in,
                              void* d_out, int out_size, void* d_ws, size_t ws_size,
                              hipStream_t stream) {
  (void)in_sizes; (void)n_in; (void)out_size; (void)ws_size;
  const float* x    = (const float*)d_in[0];
  const int*   ei   = (const int*)d_in[1];
  const float* W1   = (const float*)d_in[2];
  const float* as1w = (const float*)d_in[3];
  const float* ad1w = (const float*)d_in[4];
  const float* b1   = (const float*)d_in[5];
  const float* W2   = (const float*)d_in[6];
  const float* as2w = (const float*)d_in[7];
  const float* ad2w = (const float*)d_in[8];
  const float* b2   = (const float*)d_in[9];
  const float* ln1g = (const float*)d_in[10];
  const float* ln1b = (const float*)d_in[11];
  const float* ln2g = (const float*)d_in[12];
  const float* ln2b = (const float*)d_in[13];
  const float* decW = (const float*)d_in[14];
  const float* decb = (const float*)d_in[15];

  float* out = (float*)d_out;
  float* ws  = (float*)d_ws;

  float* h    = ws + WS_H;
  float* hn   = ws + WS_HN;
  float* agg  = ws + WS_AGG;
  float* as_  = ws + WS_AS;
  float* ad_  = ws + WS_AD;
  float* emax = ws + WS_EMAX;
  float* esum = ws + WS_ESUM;

  float* z     = out + OUT_Z;
  float* recon = out + OUT_RECON;
  float* a1    = out + OUT_A1;
  float* a2    = out + OUT_A2;

  const dim3 blk(256);
  const int gNH   = (NODES * HEADS + 255) / 256;
  const int gInit = (NODES * HIDC  + 255) / 256;
  const int gEH   = (EPR * HEADS + 255) / 256;
  const int gEC   = (EPR * HIDC  + 255) / 256;

  /* ---- Layer 1: h = x @ W1 (M=40000, K=500, N=128) ---- */
  gemm_wmma_f16<<<dim3(1, 625), blk, 0, stream>>>(x, W1, b1, h, NODES, HIDC, FIN, FIN, HIDC, HIDC, 0);
  attn_proj<<<gNH, blk, 0, stream>>>(h, as1w, ad1w, as_, ad_);
  seg_init<<<gInit, blk, 0, stream>>>(emax, esum, agg);
  edge_logits<<<gEH, blk, 0, stream>>>(ei, as_, ad_, emax, a1);
  edge_exp<<<gEH, blk, 0, stream>>>(ei, emax, esum, a1);
  alpha_norm<<<gEH, blk, 0, stream>>>(ei, esum, a1);
  edge_msg<<<gEC, blk, 0, stream>>>(ei, h, a1, agg);
  node_finalize<<<NODES / 8, blk, 0, stream>>>(agg, b1, ln1g, ln1b, hn, 0);

  /* ---- Layer 2: h = hn @ W2 (K=128) ---- */
  gemm_wmma_f16<<<dim3(1, 625), blk, 0, stream>>>(hn, W2, b2, h, NODES, HIDC, HIDC, HIDC, HIDC, HIDC, 0);
  attn_proj<<<gNH, blk, 0, stream>>>(h, as2w, ad2w, as_, ad_);
  seg_init<<<gInit, blk, 0, stream>>>(emax, esum, agg);
  edge_logits<<<gEH, blk, 0, stream>>>(ei, as_, ad_, emax, a2);
  edge_exp<<<gEH, blk, 0, stream>>>(ei, emax, esum, a2);
  alpha_norm<<<gEH, blk, 0, stream>>>(ei, esum, a2);
  edge_msg<<<gEC, blk, 0, stream>>>(ei, h, a2, agg);
  node_finalize<<<NODES / 8, blk, 0, stream>>>(agg, b2, ln2g, ln2b, z, 1);

  /* ---- Decoder: recon = z @ decW + decb (M=40000, K=128, N=500) ---- */
  gemm_wmma_f16<<<dim3(4, 625), blk, 0, stream>>>(z, decW, decb, recon, NODES, FIN, HIDC, HIDC, FIN, FIN, 1);
}
